// NerDetector_10883447128510
// MI455X (gfx1250) — matrix operand order repaired and checked by reading, not compiled, not run on hardware
//
#include <hip/hip_runtime.h>

// ---------------------------------------------------------------------------
// NER detector for MI455X (gfx1250, wave32, WMMA).
//   B=32, T=512, D=H=768, W=255, 4H=3072, NUM_NER=9
// Pipeline:
//   K1 prep : repack W_hh|W_ih -> bf16 Bpack [3072,1536] with GATE-INTERLEAVED
//             column order: wave v, tile tt (0..11) -> gate tt/3, h-cols
//             v*48 + (tt%3)*16. This puts i/f/g/o for one h-column at the same
//             (lane, VGPR) across 4 accumulators -> fully in-register LSTM cell.
//   K2 pool : first-subword gather (argmax semantics) -> pooled bf16 [8160,768]
//   K3 lstm : persistent fused recurrence, 4 workgroups (2 batch-tiles x 2 dirs)
//             512 thr = 16 waves x 12 N-tiles; per step [16 x 3072] =
//             [h_t | x_t](16x1536) @ Bpack^T via v_wmma_f32_16x16x32_bf16,
//             rotating 4-fragment B prefetch, bias pre-folded into acc,
//             cell state + gate combine entirely in registers, LDS = A buffer
//             only (48 KB), 2 barriers/step.
//   K4 head : logits = [h_f|h_b] @ W_lin^T + b_lin; softmax + argmax.
// ---------------------------------------------------------------------------

#define BATCH 32
#define TTOK  512
#define DIM   768
#define WWORDS 255
#define GATES 3072             // 4*H
#define KCAT  1536             // H + D (global Bpack row stride)
#define ASTRIDE 1544           // LDS A row stride in bf16 (+8 pad: 772 dw % 64 = 4)
#define NROWS (BATCH * WWORDS) // 8160
#define NCLS  9

typedef __attribute__((ext_vector_type(16))) __bf16 v16bf;
typedef __attribute__((ext_vector_type(8)))  __bf16 v8bf;
typedef __attribute__((ext_vector_type(8)))  float  v8f;

__device__ __forceinline__ __bf16 f2bf(float f) {
  unsigned u = __builtin_bit_cast(unsigned, f);
  unsigned r = (u + 0x7FFFu + ((u >> 16) & 1u)) >> 16;   // RNE
  unsigned short s = (unsigned short)r;
  return __builtin_bit_cast(__bf16, s);
}
__device__ __forceinline__ float bf2f(__bf16 h) {
  unsigned short s = __builtin_bit_cast(unsigned short, h);
  unsigned u = ((unsigned)s) << 16;
  return __builtin_bit_cast(float, u);
}
__device__ __forceinline__ float sigm(float x) { return 1.f / (1.f + __expf(-x)); }

// ---------------------------------------------------------------------------
// K1: pack [W_hh ; W_ih] -> bf16 Bpack[dir][cp][k] with gate-interleaved
// packed-column order cp = v*192 + tt*16 + l  (v = wave, tt = tile, l = lane):
//   gate = tt/3, original row n = gate*768 + v*48 + (tt%3)*16 + l.
// ---------------------------------------------------------------------------
__global__ void ner_prep_kernel(const float* __restrict__ Wihf,
                                const float* __restrict__ Whhf,
                                const float* __restrict__ Wihb,
                                const float* __restrict__ Whhb,
                                __bf16* __restrict__ BpackF,
                                __bf16* __restrict__ BpackB) {
  const int total = GATES * KCAT;
  int idx = blockIdx.x * 256 + threadIdx.x;
  if (idx >= 2 * total) return;
  int dir = idx / total;
  int rem = idx - dir * total;
  int cp = rem / KCAT;            // packed column 0..3071
  int k  = rem - cp * KCAT;       // 0..1535
  int v  = cp / 192;              // wave
  int t2 = cp - v * 192;
  int tt = t2 >> 4;               // tile 0..11
  int l  = t2 & 15;               // lane column
  int gate = tt / 3;              // 0=i 1=f 2=g 3=o
  int cls  = tt - gate * 3;       // h-col tile within wave
  int n = gate * DIM + v * 48 + cls * 16 + l;   // original weight row
  const float* Whh = dir ? Whhb : Whhf;
  const float* Wih = dir ? Wihb : Wihf;
  float val = (k < DIM) ? Whh[n * DIM + k] : Wih[n * DIM + (k - DIM)];
  (dir ? BpackB : BpackF)[rem] = f2bf(val);
}

// ---------------------------------------------------------------------------
// K2: first-subword pooling. One block per batch row; LDS atomicMin finds the
// first token index per word (argmax-of-equality semantics: no match -> 0).
// ---------------------------------------------------------------------------
__global__ void ner_pool_kernel(const float* __restrict__ bert,
                                const int* __restrict__ wid,
                                __bf16* __restrict__ pooled) {
  __shared__ int firstIdx[WWORDS];
  const int b = blockIdx.x;
  const int t = threadIdx.x;
  for (int i = t; i < WWORDS; i += 256) firstIdx[i] = 0x7fffffff;
  __syncthreads();
  for (int tt = t; tt < TTOK; tt += 256) {
    int w = wid[b * TTOK + tt];
    if (w >= 0 && w < WWORDS) atomicMin(&firstIdx[w], tt);
  }
  __syncthreads();
  for (int i = t; i < WWORDS; i += 256)
    if (firstIdx[i] == 0x7fffffff) firstIdx[i] = 0;   // jnp.argmax(all-false)==0
  __syncthreads();
  for (int i = t; i < WWORDS * DIM; i += 256) {
    int w = i / DIM, k = i - w * DIM;
    float v = bert[((size_t)b * TTOK + firstIdx[w]) * DIM + k];
    pooled[((size_t)(b * WWORDS + w)) * DIM + k] = f2bf(v);
  }
}

// ---------------------------------------------------------------------------
// K3: persistent fused bi-LSTM recurrence.
// grid = (2 batch-tiles, 2 directions), 512 threads = 16 waves (4/SIMD ->
// ~256-VGPR budget). LDS: Abuf [16][1544] bf16 (h | x_t, padded) = 49408 B.
// Wave v owns 12 gate-interleaved N-tiles covering h-cols [48v, 48v+48):
// 12 f32 accumulators (init = bias), 48 K-iters of v_wmma_f32_16x16x32_bf16
// with rotating 4-fragment B prefetch; LSTM cell update fully in registers.
// ---------------------------------------------------------------------------
__global__ __launch_bounds__(512) void ner_lstm_kernel(
    const __bf16* __restrict__ pooled,     // [8160][768]
    const __bf16* __restrict__ BpackF,     // [3072][1536] gate-interleaved
    const __bf16* __restrict__ BpackB,
    const float*  __restrict__ biasF,      // [3072]
    const float*  __restrict__ biasB,
    __bf16* __restrict__ houtF,            // [8160][768]
    __bf16* __restrict__ houtB) {
  extern __shared__ char smem[];
  __bf16* Abuf = (__bf16*)smem;                                    // [16][1544]

  const int bt  = blockIdx.x;        // batch tile: rows bt*16 .. bt*16+15
  const int dir = blockIdx.y;        // 0 = forward, 1 = backward
  const __bf16* Bpack = dir ? BpackB : BpackF;
  const float*  bias  = dir ? biasB : biasF;
  __bf16* hout        = dir ? houtB : houtF;

  const int tid  = threadIdx.x;
  const int lane = tid & 31;
  const int wave = tid >> 5;         // 0..15
  const int mrow = lane & 15;        // A row / B column within tile
  const int half = lane >> 4;        // lane-group select (ISA 7.12.2 layouts)

  // zero h region of Abuf (h0 = 0), 16B chunks: 16 rows x 96 chunks
  {
    uint4 z; z.x = z.y = z.z = z.w = 0u;
    for (int i = tid; i < 16 * 96; i += 512) {
      int m = i / 96, c8 = i - m * 96;
      *(uint4*)(Abuf + m * ASTRIDE + c8 * 8) = z;
    }
  }

  // cell state: 24 slots/thread = (cls 0..2) x (r 0..7) for
  // (m = r + half*8, hcol = wave*48 + cls*16 + mrow)
  float cst[24];
  #pragma unroll
  for (int p = 0; p < 24; ++p) cst[p] = 0.f;

  // per-lane bias for the 12 accumulators (acc init = bias broadcast along M)
  float bcol[12];
  #pragma unroll
  for (int tt = 0; tt < 12; ++tt) {
    int gate = tt / 3, cls = tt - gate * 3;
    bcol[tt] = bias[gate * DIM + wave * 48 + cls * 16 + mrow];
  }

  // per-lane B-fragment base: packed col wave*192 + mrow, K offset half*16
  const __bf16* bptr = Bpack + (size_t)(wave * 192 + mrow) * KCAT + half * 16;
  const __bf16* arow = Abuf + mrow * ASTRIDE;

  auto loadA = [&](int k0) -> v16bf {
    return *(const v16bf*)(arow + k0 + half * 16);
  };
  auto loadB4 = [&](v16bf* dst, int jbase, int k0) {
    #pragma unroll
    for (int j = 0; j < 4; ++j)
      dst[j] = *(const v16bf*)(bptr + (size_t)(jbase + j) * (16 * KCAT) + k0);
  };
  auto stage_x = [&](int w) {
    for (int i = tid; i < 16 * 96; i += 512) {
      int m = i / 96, c8 = i - m * 96;
      int b = bt * 16 + m;
      uint4 v = *(const uint4*)(pooled + ((size_t)(b * WWORDS + w)) * DIM + c8 * 8);
      *(uint4*)(Abuf + m * ASTRIDE + DIM + c8 * 8) = v;
    }
  };

  stage_x(dir ? (WWORDS - 1) : 0);
  __syncthreads();

  for (int s = 0; s < WWORDS; ++s) {
    const int w = dir ? (WWORDS - 1 - s) : s;

    v8f acc[12];
    #pragma unroll
    for (int tt = 0; tt < 12; ++tt)
      #pragma unroll
      for (int e = 0; e < 8; ++e) acc[tt][e] = bcol[tt];   // bias pre-folded

    auto wmma4 = [&](const v16bf& a, const v16bf* bb, int jbase) {
      #pragma unroll
      for (int j = 0; j < 4; ++j)
        acc[jbase + j] = __builtin_amdgcn_wmma_f32_16x16x32_bf16(
            false, a, false, bb[j], (short)0, acc[jbase + j], false, false);
    };

    // K loop: tiles grouped G0={0..3} G1={4..7} G2={8..11}; two rotating
    // 4-fragment buffers X/Y (64 VGPRs live), roles swap with period 2 ->
    // manual unroll by 2. Invariant at pair start: X holds G0(kk).
    v16bf bx[4], by[4];
    loadB4(bx, 0, 0);
    for (int kk = 0; kk < 46; kk += 2) {
      const int k0 = kk * 32;
      loadB4(by, 4, k0);
      v16bf a0 = loadA(k0);
      wmma4(a0, bx, 0);
      loadB4(bx, 8, k0);
      wmma4(a0, by, 4);
      loadB4(by, 0, k0 + 32);
      wmma4(a0, bx, 8);
      loadB4(bx, 4, k0 + 32);
      v16bf a1 = loadA(k0 + 32);
      wmma4(a1, by, 0);
      loadB4(by, 8, k0 + 32);
      wmma4(a1, bx, 4);
      loadB4(bx, 0, k0 + 64);       // prefetch next pair's G0
      wmma4(a1, by, 8);
    }
    {   // epilogue pair: kk = 46, 47 (no trailing prefetch)
      const int k0 = 46 * 32;
      loadB4(by, 4, k0);
      v16bf a0 = loadA(k0);
      wmma4(a0, bx, 0);
      loadB4(bx, 8, k0);
      wmma4(a0, by, 4);
      loadB4(by, 0, k0 + 32);
      wmma4(a0, bx, 8);
      loadB4(bx, 4, k0 + 32);
      v16bf a1 = loadA(k0 + 32);
      wmma4(a1, by, 0);
      loadB4(by, 8, k0 + 32);
      wmma4(a1, bx, 4);
      wmma4(a1, by, 8);
    }

    __syncthreads();   // all waves' A-fragment reads complete before h overwrite

    // in-register LSTM cell update: i/f/g/o for (m, hcol) sit at the same
    // (lane, r) in acc[cls], acc[3+cls], acc[6+cls], acc[9+cls].
    #pragma unroll
    for (int cls = 0; cls < 3; ++cls) {
      const int hcol = wave * 48 + cls * 16 + mrow;
      #pragma unroll
      for (int r = 0; r < 8; ++r) {
        float gi = acc[cls][r];
        float gf = acc[3 + cls][r];
        float gg = acc[6 + cls][r];
        float go = acc[9 + cls][r];
        float cp = cst[cls * 8 + r];
        float cn = sigm(gf) * cp + sigm(gi) * tanhf(gg);
        cst[cls * 8 + r] = cn;
        float h = sigm(go) * tanhf(cn);
        __bf16 hb16 = f2bf(h);
        const int m = r + half * 8;
        Abuf[m * ASTRIDE + hcol] = hb16;                       // next step's A
        hout[((size_t)((bt * 16 + m) * WWORDS + w)) * DIM + hcol] = hb16;
      }
    }
    // stage next step's x_t (disjoint from h region written above)
    if (s + 1 < WWORDS) stage_x(dir ? (WWORDS - 2 - s) : (s + 1));
    __syncthreads();   // h + x staged before next step's GEMM
  }
}

// ---------------------------------------------------------------------------
// K4: linear head + softmax + argmax. One block per (b,w) token, 256 threads.
// ---------------------------------------------------------------------------
__global__ void ner_head_kernel(const __bf16* __restrict__ hf,
                                const __bf16* __restrict__ hb,
                                const float* __restrict__ Wlin,   // [9][1536]
                                const float* __restrict__ blin,   // [9]
                                float* __restrict__ out) {
  __shared__ float red[NCLS * 256];
  const int r = blockIdx.x;       // 0..8159
  const int t = threadIdx.x;
  float p[NCLS];
  #pragma unroll
  for (int cc = 0; cc < NCLS; ++cc) p[cc] = 0.f;
  for (int n = t; n < 2 * DIM; n += 256) {
    float h = (n < DIM) ? bf2f(hf[(size_t)r * DIM + n])
                        : bf2f(hb[(size_t)r * DIM + (n - DIM)]);
    #pragma unroll
    for (int cc = 0; cc < NCLS; ++cc) p[cc] += h * Wlin[cc * (2 * DIM) + n];
  }
  #pragma unroll
  for (int cc = 0; cc < NCLS; ++cc) red[cc * 256 + t] = p[cc];
  __syncthreads();
  for (int st = 128; st > 0; st >>= 1) {
    if (t < st) {
      #pragma unroll
      for (int cc = 0; cc < NCLS; ++cc)
        red[cc * 256 + t] += red[cc * 256 + t + st];
    }
    __syncthreads();
  }
  if (t == 0) {
    float lg[NCLS];
    float mx = -3.402823466e+38f;
    #pragma unroll
    for (int cc = 0; cc < NCLS; ++cc) {
      lg[cc] = red[cc * 256] + blin[cc];
      mx = fmaxf(mx, lg[cc]);
    }
    int best = 0;
    float bv = lg[0];
    #pragma unroll
    for (int cc = 1; cc < NCLS; ++cc)
      if (lg[cc] > bv) { bv = lg[cc]; best = cc; }   // first max wins ties
    float se = 0.f;
    #pragma unroll
    for (int cc = 0; cc < NCLS; ++cc) { lg[cc] = expf(lg[cc] - mx); se += lg[cc]; }
    float inv = 1.f / se;
    #pragma unroll
    for (int cc = 0; cc < NCLS; ++cc) out[(size_t)r * NCLS + cc] = lg[cc] * inv;
    out[(size_t)NROWS * NCLS + r] = (float)best;     // path, in output dtype
  }
}

// ---------------------------------------------------------------------------
extern "C" void kernel_launch(void* const* d_in, const int* in_sizes, int n_in,
                              void* d_out, int out_size, void* d_ws, size_t ws_size,
                              hipStream_t stream) {
  const float* bert  = (const float*)d_in[0];
  const int*   wid   = (const int*)d_in[1];
  const float* Wihf  = (const float*)d_in[2];
  const float* Whhf  = (const float*)d_in[3];
  const float* bf    = (const float*)d_in[4];
  const float* Wihb  = (const float*)d_in[5];
  const float* Whhb  = (const float*)d_in[6];
  const float* bb    = (const float*)d_in[7];
  const float* Wlin  = (const float*)d_in[8];
  const float* blin  = (const float*)d_in[9];
  float* out = (float*)d_out;
  (void)in_sizes; (void)n_in; (void)out_size; (void)ws_size;

  // workspace layout (bf16 buffers)
  char* ws = (char*)d_ws;
  size_t o = 0;
  __bf16* BpackF = (__bf16*)(ws + o); o += (size_t)GATES * KCAT * 2;  // 9.4 MB
  __bf16* BpackB = (__bf16*)(ws + o); o += (size_t)GATES * KCAT * 2;  // 9.4 MB
  __bf16* pooled = (__bf16*)(ws + o); o += (size_t)NROWS * DIM * 2;   // 12.5 MB
  __bf16* hf     = (__bf16*)(ws + o); o += (size_t)NROWS * DIM * 2;   // 12.5 MB
  __bf16* hb     = (__bf16*)(ws + o); o += (size_t)NROWS * DIM * 2;   // 12.5 MB

  // K1: weight repack (gate-interleaved)
  {
    int total = 2 * GATES * KCAT;
    ner_prep_kernel<<<(total + 255) / 256, 256, 0, stream>>>(
        Wihf, Whhf, Wihb, Whhb, BpackF, BpackB);
  }
  // K2: pooling
  ner_pool_kernel<<<BATCH, 256, 0, stream>>>(bert, wid, pooled);

  // K3: fused bi-LSTM (48 KB dynamic LDS)
  {
    const int smem = 16 * ASTRIDE * (int)sizeof(__bf16);
    ner_lstm_kernel<<<dim3(2, 2), 512, smem, stream>>>(
        pooled, BpackF, BpackB, bf, bb, hf, hb);
  }
  // K4: head
  ner_head_kernel<<<NROWS, 256, 0, stream>>>(hf, hb, Wlin, blin, out);
}
